// DAttentionBaseline_84310208020553
// MI455X (gfx1250) — matrix operand order repaired and checked by reading, not compiled, not run on hardware
//
#include <hip/hip_runtime.h>
#include <math.h>

// ---------------------------------------------------------------------------
// Types for CDNA5 WMMA (wave32, 16x16x32 bf16 -> f32 accumulate)
// ---------------------------------------------------------------------------
typedef __bf16 bf16;
typedef __attribute__((ext_vector_type(16))) __bf16 v16bf;
typedef __attribute__((ext_vector_type(8)))  float  v8f;
typedef __attribute__((ext_vector_type(4)))  unsigned int u32x4;
typedef __attribute__((ext_vector_type(8)))  unsigned int u32x8;
typedef __attribute__((ext_vector_type(4)))  int i32x4;
typedef __attribute__((ext_vector_type(8)))  int i32x8;

#if __has_builtin(__builtin_amdgcn_tensor_load_to_lds)
#define HAVE_TDM 1
#endif

__device__ __forceinline__ v8f wmma_bf16(v16bf a, v16bf b, v8f c) {
  return __builtin_amdgcn_wmma_f32_16x16x32_bf16(false, a, false, b, (short)0, c,
                                                 false, false);
}

// A fragment (16x32 bf16, ISA layout): lane L -> row m = L&15, half = L>>4,
// K values {half*8+0..7, 16+half*8+0..7}. Two b128 loads per fragment.
__device__ __forceinline__ v16bf load_frag_a(const bf16* __restrict__ A, int lda,
                                             int mbase, int kbase, int lane) {
  const int m = mbase + (lane & 15);
  const int half = lane >> 4;
  const bf16* p = A + (size_t)m * lda + kbase + half * 8;
  u32x4 lo = *reinterpret_cast<const u32x4*>(p);
  u32x4 hi = *reinterpret_cast<const u32x4*>(p + 16);
  u32x8 w = __builtin_shufflevector(lo, hi, 0, 1, 2, 3, 4, 5, 6, 7);
  return __builtin_bit_cast(v16bf, w);
}

// A fragment from an LDS tile with padded 80B row stride (conflict-free).
__device__ __forceinline__ v16bf lds_frag_a(const unsigned char* tile, int mloc,
                                            int lane) {
  const int m = mloc + (lane & 15);
  const int half = lane >> 4;
  const unsigned char* rowp = tile + m * 80 + half * 16;
  u32x4 lo = *reinterpret_cast<const u32x4*>(rowp);
  u32x4 hi = *reinterpret_cast<const u32x4*>(rowp + 32);
  u32x8 w = __builtin_shufflevector(lo, hi, 0, 1, 2, 3, 4, 5, 6, 7);
  return __builtin_bit_cast(v16bf, w);
}

// B fragment (32x16 bf16): lane L -> col n = L&15, half = L>>4,
// K values {half*8+0..7, 16+half*8+0..7}. Bt is B-transposed ([N, K] row-major weights).
__device__ __forceinline__ v16bf load_frag_b(const bf16* __restrict__ Bt, int ldb,
                                             int nbase, int kbase, int lane) {
  const int n = nbase + (lane & 15);
  const int half = lane >> 4;
  const bf16* p = Bt + (size_t)n * ldb + kbase + half * 8;
  u32x4 lo = *reinterpret_cast<const u32x4*>(p);
  u32x4 hi = *reinterpret_cast<const u32x4*>(p + 16);
  u32x8 w = __builtin_shufflevector(lo, hi, 0, 1, 2, 3, 4, 5, 6, 7);
  return __builtin_bit_cast(v16bf, w);
}

__device__ __forceinline__ float gelu_f(float x) {
  return 0.5f * x * (1.0f + erff(x * 0.70710678118654752f));
}

#ifdef HAVE_TDM
// Issue a TDM load of a [64 rows x 32 cols] bf16 tile (row stride K elements)
// from global into LDS at lds_off, padding each 64B row with 16B (80B stride).
// D# layout per CDNA5 ISA ch.8 (group0 128b, group1 256b).
__device__ __forceinline__ void tdm_load_a_tile(const bf16* gptr, unsigned lds_off,
                                                int K, int M) {
  unsigned long long ga = (unsigned long long)(uintptr_t)gptr;
  u32x4 g0;
  g0[0] = 1u;                                        // count=1 (valid user D#)
  g0[1] = lds_off;                                   // lds_addr (bytes)
  g0[2] = (unsigned)(ga & 0xffffffffu);              // global_addr[31:0]
  g0[3] = (unsigned)((ga >> 32) & 0x01ffffffu) | (2u << 30);  // addr[56:32], type=2
  i32x8 g1;
  // data_size=1 (2B), pad_enable=1, pad_interval=3 (16 dwords), pad_amount=3 (4 dwords)
  g1[0] = (int)((1u << 16) | (1u << 20) | (3u << 22) | (3u << 25));
  g1[1] = (int)(((unsigned)K & 0xffffu) << 16);                      // tensor_dim0 lo16
  g1[2] = (int)(((unsigned)K >> 16) | (((unsigned)M & 0xffffu) << 16)); // td0 hi | td1 lo
  g1[3] = (int)(((unsigned)M >> 16) | (32u << 16));                  // td1 hi | tile_dim0=32
  g1[4] = 64;                                                        // tile_dim1=64, tile_dim2=0
  g1[5] = K;                                                         // dim0_stride[31:0]
  g1[6] = 0;                                                         // dim0_stride hi, dim1_stride lo
  g1[7] = 0;
  i32x4 z4 = {0, 0, 0, 0};
#if __has_include(<hip/amd_detail/amd_gfx1250_TDM.h>)
  i32x8 z8 = {0, 0, 0, 0, 0, 0, 0, 0};
  __builtin_amdgcn_tensor_load_to_lds(g0, g1, z4, z4, z8, 0);
#else
  __builtin_amdgcn_tensor_load_to_lds(g0, g1, z4, z4, 0);
#endif
}
#endif

// ---------------------------------------------------------------------------
// Generic GEMM:  C[M,N] = A[M,K](bf16) * W[N,K]^T(bf16) + bias + residual
// Block = 4 waves, 64x64 block tile, wave tile 32x32 (2x2 WMMA tiles).
// A tile staged in LDS by the Tensor Data Mover (double buffered, TENSORcnt);
// B fragments read direct (weights stay L2-resident) with global_prefetch.
// out_mode 0: C row-major [M,N] f32.  out_mode 1: C[(m/1024)*N+n][m%1024] (NCHW).
// ---------------------------------------------------------------------------
__global__ __launch_bounds__(128) void gemm_bf16_k(
    const bf16* __restrict__ A, const bf16* __restrict__ Bt,
    const float* __restrict__ bias, const float* __restrict__ residual,
    float* __restrict__ C, int M, int N, int K, int out_mode) {
  const int lane = threadIdx.x & 31;
  const int wave = threadIdx.x >> 5;
  const int mloc = (wave >> 1) * 32;
  const int mbase = blockIdx.x * 64 + mloc;
  const int nbase = blockIdx.y * 64 + (wave & 1) * 32;

  v8f acc00 = {}, acc01 = {}, acc10 = {}, acc11 = {};

#ifdef HAVE_TDM
  __shared__ alignas(16) unsigned char atile[2][64 * 80];
  const bf16* Ablk = A + (size_t)blockIdx.x * 64 * K;
  if (wave == 0) {
    tdm_load_a_tile(Ablk, (unsigned)(uintptr_t)&atile[0][0], K, M);
    __builtin_amdgcn_s_wait_tensorcnt(0);
  }
  __syncthreads();
  int cur = 0;
  for (int k = 0; k < K; k += 32) {
    const bool more = (k + 32 < K);
    if (more && wave == 0)
      tdm_load_a_tile(Ablk + k + 32, (unsigned)(uintptr_t)&atile[cur ^ 1][0], K, M);
    v16bf a0 = lds_frag_a(atile[cur], mloc, lane);
    v16bf a1 = lds_frag_a(atile[cur], mloc + 16, lane);
    v16bf b0 = load_frag_b(Bt, K, nbase, k, lane);
    v16bf b1 = load_frag_b(Bt, K, nbase + 16, k, lane);
    if (more)  // CDNA5 global_prefetch_b8 of next B tile
      __builtin_prefetch(Bt + (size_t)(nbase + (lane & 15)) * K + k + 32, 0, 1);
    acc00 = wmma_bf16(a0, b0, acc00);
    acc01 = wmma_bf16(a0, b1, acc01);
    acc10 = wmma_bf16(a1, b0, acc10);
    acc11 = wmma_bf16(a1, b1, acc11);
    if (more && wave == 0) __builtin_amdgcn_s_wait_tensorcnt(0);
    __syncthreads();
    cur ^= 1;
  }
#else
  for (int k = 0; k < K; k += 32) {
    v16bf a0 = load_frag_a(A, K, mbase, k, lane);
    v16bf a1 = load_frag_a(A, K, mbase + 16, k, lane);
    v16bf b0 = load_frag_b(Bt, K, nbase, k, lane);
    v16bf b1 = load_frag_b(Bt, K, nbase + 16, k, lane);
    if (k + 32 < K) {
      __builtin_prefetch(A + (size_t)(mbase + (lane & 15)) * K + k + 32, 0, 1);
      __builtin_prefetch(Bt + (size_t)(nbase + (lane & 15)) * K + k + 32, 0, 1);
    }
    acc00 = wmma_bf16(a0, b0, acc00);
    acc01 = wmma_bf16(a0, b1, acc01);
    acc10 = wmma_bf16(a1, b0, acc10);
    acc11 = wmma_bf16(a1, b1, acc11);
  }
#endif

  const int ln16 = lane & 15, half = lane >> 4;
  v8f accs[2][2] = {{acc00, acc01}, {acc10, acc11}};
#pragma unroll
  for (int ti = 0; ti < 2; ++ti)
#pragma unroll
    for (int tj = 0; tj < 2; ++tj) {
      int ncol = nbase + tj * 16 + ln16;
      float bv = bias ? bias[ncol] : 0.f;
#pragma unroll
      for (int r = 0; r < 8; ++r) {
        int m = mbase + ti * 16 + r + 8 * half;
        float v = accs[ti][tj][r] + bv;
        if (residual) v += residual[(size_t)m * N + ncol];
        if (out_mode == 0) C[(size_t)m * N + ncol] = v;
        else { int b = m >> 10, s = m & 1023; C[(((size_t)b * N + ncol) << 10) + s] = v; }
      }
    }
}

// ---------------------------------------------------------------------------
// Fused flash-style attention.  Q:[bh,nq,D] bf16, Kt:[bh,nk,D] bf16,
// Vt:[bh,D,nk] bf16 -> O written bf16 at ((bg*nq+m)*heads + h)*D + c (head-merge).
// One wave owns 16 query rows; loop over 32-key chunks, online softmax,
// P tile converted C-layout -> A-layout via LDS.
// ---------------------------------------------------------------------------
template <int D>
__global__ __launch_bounds__(128) void attn_fused_k(
    const bf16* __restrict__ Q, const bf16* __restrict__ Kt,
    const bf16* __restrict__ Vt, bf16* __restrict__ O,
    int nq, int nk, int heads, float scale) {
  __shared__ float pbuf[4][16][36];
  const int lane = threadIdx.x & 31;
  const int wave = threadIdx.x >> 5;
  const int bh = blockIdx.y;
  const int mbase = blockIdx.x * 64 + wave * 16;
  const int ln16 = lane & 15, half = lane >> 4;

  const bf16* Qb = Q + (size_t)bh * nq * D;
  const bf16* Kb = Kt + (size_t)bh * nk * D;
  const bf16* Vb = Vt + (size_t)bh * D * nk;

  constexpr int KS = D / 32;  // k-steps of QK^T
  constexpr int NT = D / 16;  // output dim tiles
  v16bf qf[KS];
#pragma unroll
  for (int d = 0; d < KS; ++d) qf[d] = load_frag_a(Qb, D, mbase, d * 32, lane);

  v8f oacc[NT] = {};
  float rmax[8], rsum[8];
#pragma unroll
  for (int r = 0; r < 8; ++r) { rmax[r] = -1e30f; rsum[r] = 0.f; }

  for (int kc = 0; kc < nk; kc += 32) {
    v8f s0 = {}, s1 = {};
#pragma unroll
    for (int d = 0; d < KS; ++d) {
      v16bf kb0 = load_frag_b(Kb, D, kc, d * 32, lane);
      v16bf kb1 = load_frag_b(Kb, D, kc + 16, d * 32, lane);
      s0 = wmma_bf16(qf[d], kb0, s0);
      s1 = wmma_bf16(qf[d], kb1, s1);
    }
    // online softmax per row (rows live across the 16 lanes of one half-wave)
#pragma unroll
    for (int r = 0; r < 8; ++r) {
      float v0 = s0[r] * scale, v1 = s1[r] * scale;
      float mx = fmaxf(v0, v1);
#pragma unroll
      for (int off = 1; off < 16; off <<= 1) mx = fmaxf(mx, __shfl_xor(mx, off, 32));
      float nm = fmaxf(rmax[r], mx);
      float corr = __expf(rmax[r] - nm);
      float p0 = __expf(v0 - nm);
      float p1 = __expf(v1 - nm);
      float ps = p0 + p1;
#pragma unroll
      for (int off = 1; off < 16; off <<= 1) ps += __shfl_xor(ps, off, 32);
      rsum[r] = rsum[r] * corr + ps;
      rmax[r] = nm;
#pragma unroll
      for (int t = 0; t < NT; ++t) oacc[t][r] *= corr;
      pbuf[wave][r + 8 * half][ln16] = p0;
      pbuf[wave][r + 8 * half][16 + ln16] = p1;
    }
    __syncthreads();
    // read P back in A-fragment order and convert to bf16
    v16bf pa;
    {
      const float* row = &pbuf[wave][ln16][0];
#pragma unroll
      for (int e = 0; e < 8; ++e) pa[e] = (bf16)row[half * 8 + e];
#pragma unroll
      for (int e = 0; e < 8; ++e) pa[8 + e] = (bf16)row[16 + half * 8 + e];
    }
#pragma unroll
    for (int t = 0; t < NT; ++t) {
      v16bf vb = load_frag_b(Vb, nk, t * 16, kc, lane);
      oacc[t] = wmma_bf16(pa, vb, oacc[t]);
    }
    __syncthreads();
  }

  const int bg = bh / heads, h = bh % heads;
#pragma unroll
  for (int t = 0; t < NT; ++t) {
    int c = t * 16 + ln16;
#pragma unroll
    for (int r = 0; r < 8; ++r) {
      int m = mbase + r + 8 * half;
      float v = oacc[t][r] / rsum[r];
      O[(((size_t)bg * nq + m) * heads + h) * D + c] = (bf16)v;
    }
  }
}

// ---------------------------------------------------------------------------
// Small VALU kernels
// ---------------------------------------------------------------------------
__global__ void cvt_bf16_k(const float* __restrict__ in, bf16* __restrict__ out,
                           size_t n) {
  size_t i = (size_t)blockIdx.x * blockDim.x + threadIdx.x;
  if (i < n) out[i] = (bf16)in[i];
}

// feat NCHW f32 -> token-major bf16 [(b,s), c]
__global__ void nchw_tok_bf16_k(const float* __restrict__ in, bf16* __restrict__ out,
                                int Bn, int Cn, int Sn) {
  size_t i = (size_t)blockIdx.x * blockDim.x + threadIdx.x;
  size_t total = (size_t)Bn * Cn * Sn;
  if (i >= total) return;
  int c = (int)(i % Cn); size_t t = i / Cn;
  int s = (int)(t % Sn); int b = (int)(t / Sn);
  out[i] = (bf16)in[((size_t)b * Cn + c) * Sn + s];
}

// emb_w [512,1024,3,3] -> Wk bf16 [512, 9216] with k = tap*1024 + ci
__global__ void repack_embw_k(const float* __restrict__ w, bf16* __restrict__ out) {
  size_t i = (size_t)blockIdx.x * blockDim.x + threadIdx.x;
  if (i >= (size_t)512 * 9216) return;
  int k = (int)(i % 9216); int o = (int)(i / 9216);
  int tap = k >> 10, ci = k & 1023;
  int ty = tap / 3, tx = tap % 3;
  out[i] = (bf16)w[(((size_t)o * 1024 + ci) * 3 + ty) * 3 + tx];
}

// im2col for 3x3 pad-1 conv over concat(tgt, ref): col bf16 [4096, 9216]
__global__ void im2col_k(const float* __restrict__ tgt, const float* __restrict__ ref,
                         bf16* __restrict__ col) {
  size_t i = (size_t)blockIdx.x * blockDim.x + threadIdx.x;
  if (i >= (size_t)4096 * 9216) return;
  int k = (int)(i % 9216); size_t row = i / 9216;
  int n = (int)(row & 1023); int b = (int)(row >> 10);
  int tap = k >> 10, ci = k & 1023;
  int y = (n >> 5) + tap / 3 - 1;
  int x = (n & 31) + tap % 3 - 1;
  float v = 0.f;
  if (y >= 0 && y < 32 && x >= 0 && x < 32) {
    const float* src = (ci < 512) ? tgt : ref;
    v = src[(((size_t)b * 512 + (ci & 511)) << 10) + (y << 5) + x];
  }
  col[i] = (bf16)v;
}

// fg[(b*2+g)*1024+n][c] = feattok[(b*1024+n)*512 + g*256 + c]
__global__ void extract_fg_k(const float* __restrict__ feattok, float* __restrict__ fg) {
  size_t i = (size_t)blockIdx.x * blockDim.x + threadIdx.x;
  if (i >= (size_t)8192 * 256) return;
  int c = (int)(i & 255); size_t t = i >> 8;
  int n = (int)(t & 1023); int bg = (int)(t >> 10);
  int b = bg >> 1, g = bg & 1;
  fg[i] = feattok[(((size_t)(b << 10) + n) << 9) + (g << 8) + c];
}

// LayerNorm over D=256 (one wave per row); optional GELU; f32 and/or bf16 out.
__global__ void ln256_k(const float* __restrict__ in, const float* __restrict__ gw,
                        const float* __restrict__ bw, float eps, int nrows,
                        float* __restrict__ of32, bf16* __restrict__ obf, int do_gelu) {
  int lane = threadIdx.x & 31, wave = threadIdx.x >> 5;
  int row = blockIdx.x * (blockDim.x >> 5) + wave;
  if (row >= nrows) return;
  const float* p = in + (size_t)row * 256;
  float v[8]; float s = 0.f;
#pragma unroll
  for (int i = 0; i < 8; ++i) { v[i] = p[lane + i * 32]; s += v[i]; }
#pragma unroll
  for (int off = 1; off < 32; off <<= 1) s += __shfl_xor(s, off, 32);
  float mean = s * (1.f / 256.f);
  float q = 0.f;
#pragma unroll
  for (int i = 0; i < 8; ++i) { float d = v[i] - mean; q += d * d; }
#pragma unroll
  for (int off = 1; off < 32; off <<= 1) q += __shfl_xor(q, off, 32);
  float rstd = rsqrtf(q * (1.f / 256.f) + eps);
#pragma unroll
  for (int i = 0; i < 8; ++i) {
    int c = lane + i * 32;
    float y = (v[i] - mean) * rstd * gw[c] + bw[c];
    if (do_gelu) y = gelu_f(y);
    if (of32) of32[(size_t)row * 256 + c] = y;
    if (obf) obf[(size_t)row * 256 + c] = (bf16)y;
  }
}

// out[((bg*heads+h)*N + n)*d + c] = in[(bg*N+n)*ncols + coff + h*d + c]
__global__ void split_heads_k(const float* __restrict__ in, bf16* __restrict__ out,
                              int heads, int N, int d, int ncols, int coff,
                              size_t total) {
  size_t i = (size_t)blockIdx.x * blockDim.x + threadIdx.x;
  if (i >= total) return;
  int c = (int)(i % d); size_t r = i / d;
  int n = (int)(r % N); size_t r2 = r / N;
  int h = (int)(r2 % heads); int bg = (int)(r2 / heads);
  out[i] = (bf16)in[((size_t)bg * N + n) * ncols + coff + h * d + c];
}

// out[((bg*heads+h)*d + c)*N + n]  (V transposed for B-fragment loads)
__global__ void split_heads_T_k(const float* __restrict__ in, bf16* __restrict__ out,
                                int heads, int N, int d, int ncols, int coff,
                                size_t total) {
  size_t i = (size_t)blockIdx.x * blockDim.x + threadIdx.x;
  if (i >= total) return;
  int n = (int)(i % N); size_t r = i / N;
  int c = (int)(r % d); size_t r2 = r / d;
  int h = (int)(r2 % heads); int bg = (int)(r2 / heads);
  out[i] = (bf16)in[((size_t)bg * N + n) * ncols + coff + h * d + c];
}

// depthwise 3x3 pad-1 over [BG=8, hid=1024, 32, 32] in token layout + GELU -> bf16
__global__ void dw3_gelu_k(const float* __restrict__ m, const float* __restrict__ w,
                           const float* __restrict__ bias, bf16* __restrict__ out) {
  size_t i = (size_t)blockIdx.x * blockDim.x + threadIdx.x;
  if (i >= (size_t)8192 * 1024) return;
  int ch = (int)(i & 1023); size_t t = i >> 10;
  int n = (int)(t & 1023); int bg = (int)(t >> 10);
  int y = n >> 5, x = n & 31;
  float acc = bias[ch];
#pragma unroll
  for (int ty = 0; ty < 3; ++ty)
#pragma unroll
    for (int tx = 0; tx < 3; ++tx) {
      int iy = y + ty - 1, ix = x + tx - 1;
      if (iy < 0 || iy > 31 || ix < 0 || ix > 31) continue;
      acc += m[((((size_t)bg << 10) + (iy << 5) + ix) << 10) + ch] * w[ch * 9 + ty * 3 + tx];
    }
  out[i] = (bf16)gelu_f(acc);
}

// depthwise 9x9 pad-4 over [BG=8, gc=256, 32, 32] in token layout -> f32 tokens
__global__ void dw9_k(const float* __restrict__ xin, const float* __restrict__ w,
                      const float* __restrict__ bias, float* __restrict__ out) {
  size_t i = (size_t)blockIdx.x * blockDim.x + threadIdx.x;
  if (i >= (size_t)8192 * 256) return;
  int c = (int)(i & 255); size_t t = i >> 8;
  int n = (int)(t & 1023); int bg = (int)(t >> 10);
  int y = n >> 5, x = n & 31;
  float acc = bias[c];
  for (int ty = 0; ty < 9; ++ty) {
    int iy = y + ty - 4;
    if (iy < 0 || iy > 31) continue;
    for (int tx = 0; tx < 9; ++tx) {
      int ix = x + tx - 4;
      if (ix < 0 || ix > 31) continue;
      acc += xin[((((size_t)bg << 10) + (iy << 5) + ix) << 8) + c] * w[c * 81 + ty * 9 + tx];
    }
  }
  out[i] = acc;
}

// co_out (2x256 dot) + tanh offset scaling + reference grid -> pos (px, py)
__global__ void co_pos_k(const float* __restrict__ act, const float* __restrict__ w2,
                         float* __restrict__ pos, int nrows) {
  int lane = threadIdx.x & 31, wave = threadIdx.x >> 5;
  int row = blockIdx.x * (blockDim.x >> 5) + wave;
  if (row >= nrows) return;
  const float* p = act + (size_t)row * 256;
  float s0 = 0.f, s1 = 0.f;
#pragma unroll
  for (int i = 0; i < 8; ++i) {
    int c = lane + i * 32;
    float a = p[c];
    s0 += a * w2[c];
    s1 += a * w2[256 + c];
  }
#pragma unroll
  for (int off = 1; off < 32; off <<= 1) {
    s0 += __shfl_xor(s0, off, 32);
    s1 += __shfl_xor(s1, off, 32);
  }
  if (lane == 0) {
    int n = row & 1023; int y = n >> 5, x = n & 31;
    float offy = tanhf(s0) * (2.0f / 32.0f);
    float offx = tanhf(s1) * (2.0f / 32.0f);
    float refy = ((y + 0.5f) / 32.0f) * 2.f - 1.f;
    float refx = ((x + 0.5f) / 32.0f) * 2.f - 1.f;
    pos[(size_t)row * 2 + 0] = offx + refx;
    pos[(size_t)row * 2 + 1] = offy + refy;
  }
}

// bilinear grid sample (align_corners=True, zero pad) -> sampled tokens bf16
__global__ void gsample_k(const float* __restrict__ feattok, const float* __restrict__ pos,
                          bf16* __restrict__ out) {
  size_t i = (size_t)blockIdx.x * blockDim.x + threadIdx.x;
  if (i >= (size_t)8192 * 256) return;
  int c = (int)(i & 255); size_t t = i >> 8;
  int n = (int)(t & 1023); int bg = (int)(t >> 10);
  int b = bg >> 1, g = bg & 1, ch = (g << 8) + c;
  float px = pos[t * 2], py = pos[t * 2 + 1];
  float fx = (px + 1.f) * 0.5f * 31.f;
  float fy = (py + 1.f) * 0.5f * 31.f;
  float x0 = floorf(fx), y0 = floorf(fy);
  float wx1 = fx - x0, wx0 = 1.f - wx1;
  float wy1 = fy - y0, wy0 = 1.f - wy1;
  int ix0 = (int)x0, iy0 = (int)y0;
  auto tap = [&](int ix, int iy) -> float {
    if (ix < 0 || ix > 31 || iy < 0 || iy > 31) return 0.f;
    return feattok[(((size_t)(b << 10) + (iy << 5) + ix) << 9) + ch];
  };
  float r = tap(ix0, iy0) * wx0 * wy0 + tap(ix0 + 1, iy0) * wx1 * wy0 +
            tap(ix0, iy0 + 1) * wx0 * wy1 + tap(ix0 + 1, iy0 + 1) * wx1 * wy1;
  out[(((size_t)(b << 10) + n) << 9) + ch] = (bf16)r;
}

// ---------------------------------------------------------------------------
// Host orchestration
// ---------------------------------------------------------------------------
#define CDIV(a, b) (((a) + (b)-1) / (b))

extern "C" void kernel_launch(void* const* d_in, const int* in_sizes, int n_in,
                              void* d_out, int out_size, void* d_ws, size_t ws_size,
                              hipStream_t stream) {
  const float* feat_tgt = (const float*)d_in[0];
  const float* feat_ref = (const float*)d_in[1];
  const float* emb_w = (const float*)d_in[2];
  const float* emb_b = (const float*)d_in[3];
  const float* q_w = (const float*)d_in[4];
  const float* q_b = (const float*)d_in[5];
  const float* k_w = (const float*)d_in[6];
  const float* k_b = (const float*)d_in[7];
  const float* v_w = (const float*)d_in[8];
  const float* v_b = (const float*)d_in[9];
  const float* out_w = (const float*)d_in[10];
  const float* out_b = (const float*)d_in[11];
  const float* ln1_w = (const float*)d_in[12];
  const float* ln1_b = (const float*)d_in[13];
  const float* n1_w = (const float*)d_in[14];
  const float* n1_b = (const float*)d_in[15];
  const float* mbq_w = (const float*)d_in[16];
  const float* mbq_b = (const float*)d_in[17];
  const float* mbkv_w = (const float*)d_in[18];
  const float* mbkv_b = (const float*)d_in[19];
  const float* mbp_w = (const float*)d_in[20];
  const float* mbp_b = (const float*)d_in[21];
  const float* n2_w = (const float*)d_in[22];
  const float* n2_b = (const float*)d_in[23];
  const float* fc1_w = (const float*)d_in[24];
  const float* fc1_b = (const float*)d_in[25];
  const float* dw_w = (const float*)d_in[26];
  const float* dw_b = (const float*)d_in[27];
  const float* fc2_w = (const float*)d_in[28];
  const float* fc2_b = (const float*)d_in[29];
  const float* codw_w = (const float*)d_in[30];
  const float* codw_b = (const float*)d_in[31];
  const float* coln_w = (const float*)d_in[32];
  const float* coln_b = (const float*)d_in[33];
  const float* coout_w = (const float*)d_in[34];
  float* y_out = (float*)d_out;

  // bump allocator on workspace (256B aligned)
  char* wp = (char*)d_ws;
  auto alloc = [&](size_t bytes) -> void* {
    void* p = (void*)wp;
    wp += (bytes + 255) & ~(size_t)255;
    return p;
  };

  // bf16 weights
  bf16* wEmb = (bf16*)alloc((size_t)512 * 9216 * 2);
  bf16* wQ = (bf16*)alloc((size_t)512 * 512 * 2);
  bf16* wK = (bf16*)alloc((size_t)512 * 512 * 2);
  bf16* wV = (bf16*)alloc((size_t)512 * 512 * 2);
  bf16* wO = (bf16*)alloc((size_t)512 * 512 * 2);
  bf16* wMbQ = (bf16*)alloc((size_t)256 * 256 * 2);
  bf16* wMbKv = (bf16*)alloc((size_t)512 * 256 * 2);
  bf16* wMbP = (bf16*)alloc((size_t)256 * 256 * 2);
  bf16* wFc1 = (bf16*)alloc((size_t)1024 * 256 * 2);
  bf16* wFc2 = (bf16*)alloc((size_t)256 * 1024 * 2);
  // activations
  bf16* aTgt = (bf16*)alloc((size_t)4096 * 512 * 2);
  bf16* colB = (bf16*)alloc((size_t)4096 * 9216 * 2);
  float* feattok = (float*)alloc((size_t)4096 * 512 * 4);
  float* qtok = (float*)alloc((size_t)4096 * 512 * 4);
  bf16* qMain = (bf16*)alloc((size_t)32 * 1024 * 64 * 2);
  float* fg = (float*)alloc((size_t)8192 * 256 * 4);
  float* x1 = (float*)alloc((size_t)8192 * 256 * 4);
  bf16* hbf = (bf16*)alloc((size_t)8192 * 256 * 2);
  float* qm = (float*)alloc((size_t)8192 * 256 * 4);
  bf16* qMit = (bf16*)alloc((size_t)64 * 1024 * 32 * 2);
  float* kvm = (float*)alloc((size_t)8192 * 512 * 4);
  bf16* kMit = (bf16*)alloc((size_t)64 * 1024 * 32 * 2);
  bf16* vtMit = (bf16*)alloc((size_t)64 * 32 * 1024 * 2);
  bf16* oMit = (bf16*)alloc((size_t)8192 * 256 * 2);
  float* x2 = (float*)alloc((size_t)8192 * 256 * 4);
  bf16* h2bf = (bf16*)alloc((size_t)8192 * 256 * 2);
  float* mfc = (float*)alloc((size_t)8192 * 1024 * 4);
  bf16* mbf = (bf16*)alloc((size_t)8192 * 1024 * 2);
  float* x3 = (float*)alloc((size_t)8192 * 256 * 4);
  float* cof = (float*)alloc((size_t)8192 * 256 * 4);
  float* coact = (float*)alloc((size_t)8192 * 256 * 4);
  float* pos = (float*)alloc((size_t)8192 * 2 * 4);
  bf16* sampB = (bf16*)alloc((size_t)4096 * 512 * 2);
  float* ktok = (float*)alloc((size_t)4096 * 512 * 4);
  float* vtok = (float*)alloc((size_t)4096 * 512 * 4);
  bf16* kMain = (bf16*)alloc((size_t)32 * 1024 * 64 * 2);
  bf16* vtMain = (bf16*)alloc((size_t)32 * 64 * 1024 * 2);
  bf16* attB = (bf16*)alloc((size_t)4096 * 512 * 2);

  const int TB = 256;

  // ---- weight conversions ----
  cvt_bf16_k<<<CDIV(512 * 512, TB), TB, 0, stream>>>(q_w, wQ, (size_t)512 * 512);
  cvt_bf16_k<<<CDIV(512 * 512, TB), TB, 0, stream>>>(k_w, wK, (size_t)512 * 512);
  cvt_bf16_k<<<CDIV(512 * 512, TB), TB, 0, stream>>>(v_w, wV, (size_t)512 * 512);
  cvt_bf16_k<<<CDIV(512 * 512, TB), TB, 0, stream>>>(out_w, wO, (size_t)512 * 512);
  cvt_bf16_k<<<CDIV(256 * 256, TB), TB, 0, stream>>>(mbq_w, wMbQ, (size_t)256 * 256);
  cvt_bf16_k<<<CDIV(512 * 256, TB), TB, 0, stream>>>(mbkv_w, wMbKv, (size_t)512 * 256);
  cvt_bf16_k<<<CDIV(256 * 256, TB), TB, 0, stream>>>(mbp_w, wMbP, (size_t)256 * 256);
  cvt_bf16_k<<<CDIV(1024 * 256, TB), TB, 0, stream>>>(fc1_w, wFc1, (size_t)1024 * 256);
  cvt_bf16_k<<<CDIV(256 * 1024, TB), TB, 0, stream>>>(fc2_w, wFc2, (size_t)256 * 1024);
  repack_embw_k<<<CDIV(512 * 9216, TB), TB, 0, stream>>>(emb_w, wEmb);
  nchw_tok_bf16_k<<<CDIV(4 * 512 * 1024, TB), TB, 0, stream>>>(feat_tgt, aTgt, 4, 512, 1024);

  // ---- embedding conv as im2col GEMM: feattok[(b,n), 512] ----
  im2col_k<<<(unsigned)CDIV((size_t)4096 * 9216, TB), TB, 0, stream>>>(feat_tgt, feat_ref, colB);
  gemm_bf16_k<<<dim3(64, 8), 128, 0, stream>>>(colB, wEmb, emb_b, nullptr, feattok,
                                               4096, 512, 9216, 0);

  // ---- q projection + head split ----
  gemm_bf16_k<<<dim3(64, 8), 128, 0, stream>>>(aTgt, wQ, q_b, nullptr, qtok,
                                               4096, 512, 512, 0);
  split_heads_k<<<CDIV(2097152, TB), TB, 0, stream>>>(qtok, qMain, 8, 1024, 64, 512, 0,
                                                      (size_t)2097152);

  // ---- group tokens + LN1 -> x1 ----
  extract_fg_k<<<CDIV(2097152, TB), TB, 0, stream>>>(feattok, fg);
  ln256_k<<<1024, 256, 0, stream>>>(fg, ln1_w, ln1_b, 1e-5f, 8192, x1, nullptr, 0);

  // ---- MitBlock attention ----
  ln256_k<<<1024, 256, 0, stream>>>(x1, n1_w, n1_b, 1e-6f, 8192, nullptr, hbf, 0);
  gemm_bf16_k<<<dim3(128, 4), 128, 0, stream>>>(hbf, wMbQ, mbq_b, nullptr, qm,
                                                8192, 256, 256, 0);
  gemm_bf16_k<<<dim3(128, 8), 128, 0, stream>>>(hbf, wMbKv, mbkv_b, nullptr, kvm,
                                                8192, 512, 256, 0);
  split_heads_k<<<CDIV(2097152, TB), TB, 0, stream>>>(qm, qMit, 8, 1024, 32, 256, 0,
                                                      (size_t)2097152);
  split_heads_k<<<CDIV(2097152, TB), TB, 0, stream>>>(kvm, kMit, 8, 1024, 32, 512, 0,
                                                      (size_t)2097152);
  split_heads_T_k<<<CDIV(2097152, TB), TB, 0, stream>>>(kvm, vtMit, 8, 1024, 32, 512, 256,
                                                        (size_t)2097152);
  attn_fused_k<32><<<dim3(16, 64), 128, 0, stream>>>(qMit, kMit, vtMit, oMit,
                                                     1024, 1024, 8, 0.17677669529663689f);
  gemm_bf16_k<<<dim3(128, 4), 128, 0, stream>>>(oMit, wMbP, mbp_b, x1, x2,
                                                8192, 256, 256, 0);

  // ---- MitBlock MLP (fc1 -> dw3x3 -> gelu -> fc2) ----
  ln256_k<<<1024, 256, 0, stream>>>(x2, n2_w, n2_b, 1e-6f, 8192, nullptr, h2bf, 0);
  gemm_bf16_k<<<dim3(128, 16), 128, 0, stream>>>(h2bf, wFc1, fc1_b, nullptr, mfc,
                                                 8192, 1024, 256, 0);
  dw3_gelu_k<<<(unsigned)CDIV((size_t)8192 * 1024, TB), TB, 0, stream>>>(mfc, dw_w, dw_b, mbf);
  gemm_bf16_k<<<dim3(128, 4), 128, 0, stream>>>(mbf, wFc2, fc2_b, x2, x3,
                                                8192, 256, 1024, 0);

  // ---- conv_offset stack -> sampling positions ----
  dw9_k<<<CDIV(2097152, TB), TB, 0, stream>>>(x3, codw_w, codw_b, cof);
  ln256_k<<<1024, 256, 0, stream>>>(cof, coln_w, coln_b, 1e-5f, 8192, coact, nullptr, 1);
  co_pos_k<<<1024, 256, 0, stream>>>(coact, coout_w, pos, 8192);

  // ---- bilinear grid sample of feat -> sampled tokens ----
  gsample_k<<<CDIV(2097152, TB), TB, 0, stream>>>(feattok, pos, sampB);

  // ---- K/V projections + head split ----
  gemm_bf16_k<<<dim3(64, 8), 128, 0, stream>>>(sampB, wK, k_b, nullptr, ktok,
                                               4096, 512, 512, 0);
  gemm_bf16_k<<<dim3(64, 8), 128, 0, stream>>>(sampB, wV, v_b, nullptr, vtok,
                                               4096, 512, 512, 0);
  split_heads_k<<<CDIV(2097152, TB), TB, 0, stream>>>(ktok, kMain, 8, 1024, 64, 512, 0,
                                                      (size_t)2097152);
  split_heads_T_k<<<CDIV(2097152, TB), TB, 0, stream>>>(vtok, vtMain, 8, 1024, 64, 512, 0,
                                                        (size_t)2097152);

  // ---- main deformable attention + output projection (NCHW store) ----
  attn_fused_k<64><<<dim3(16, 32), 128, 0, stream>>>(qMain, kMain, vtMain, attB,
                                                     1024, 1024, 8, 0.125f);
  gemm_bf16_k<<<dim3(64, 8), 128, 0, stream>>>(attB, wO, out_b, nullptr, y_out,
                                               4096, 512, 512, 1);
}
